// MultiHeadedAttention_83932250898499
// MI455X (gfx1250) — compile-verified, matrix operand order checked by reading, not run on hardware
//
#include <hip/hip_runtime.h>

typedef __bf16 bf16;
typedef __attribute__((ext_vector_type(16))) __bf16 v16bf;
typedef __attribute__((ext_vector_type(8)))  __bf16 v8bf;
typedef __attribute__((ext_vector_type(8)))  float  v8f;
typedef __attribute__((ext_vector_type(4)))  unsigned int u32x4;
typedef __attribute__((ext_vector_type(8)))  int i32x8;
typedef __attribute__((ext_vector_type(4)))  int i32x4;

#define HEADS   16
#define D_MODEL 1024
#define D_K     64
#define SEQ     2048
#define BATCH   4

#if __has_builtin(__builtin_amdgcn_tensor_load_to_lds) && __has_builtin(__builtin_amdgcn_s_wait_tensorcnt)
#define HAVE_TDM 1
#else
#define HAVE_TDM 0
#endif

// Build a 16-element bf16 A/B fragment from two contiguous 16-byte chunks.
// CDNA5 16-bit A-layout: lane holds row M=lane%16; elements 0..7 are
// K = kbase + (lane/16)*8 + e, elements 8..15 are K = kbase+16 + (lane/16)*8 + e.
__device__ __forceinline__ v16bf load_frag(const bf16* lo, const bf16* hi) {
    v8bf a = *(const v8bf*)lo;
    v8bf b = *(const v8bf*)hi;
    v16bf r;
#pragma unroll
    for (int i = 0; i < 8; ++i) { r[i] = a[i]; r[i + 8] = b[i]; }
    return r;
}

// Fused DPP butterfly steps: one VOP2+DPP instruction per step
// (avoids the mov_dpp + canonicalize pair the intrinsic path emits).
__device__ __forceinline__ float red_max16(float x) {
    float t;
    asm("v_max_num_f32_dpp %0, %1, %1 quad_perm:[1,0,3,2] row_mask:0xf bank_mask:0xf bound_ctrl:1"
        : "=&v"(t) : "v"(x));
    x = t;
    asm("v_max_num_f32_dpp %0, %1, %1 quad_perm:[2,3,0,1] row_mask:0xf bank_mask:0xf bound_ctrl:1"
        : "=&v"(t) : "v"(x));
    x = t;
    asm("v_max_num_f32_dpp %0, %1, %1 row_half_mirror row_mask:0xf bank_mask:0xf bound_ctrl:1"
        : "=&v"(t) : "v"(x));
    x = t;
    asm("v_max_num_f32_dpp %0, %1, %1 row_mirror row_mask:0xf bank_mask:0xf bound_ctrl:1"
        : "=&v"(t) : "v"(x));
    return t;
}
__device__ __forceinline__ float red_sum16(float x) {
    float t;
    asm("v_add_f32_dpp %0, %1, %1 quad_perm:[1,0,3,2] row_mask:0xf bank_mask:0xf bound_ctrl:1"
        : "=&v"(t) : "v"(x));
    x = t;
    asm("v_add_f32_dpp %0, %1, %1 quad_perm:[2,3,0,1] row_mask:0xf bank_mask:0xf bound_ctrl:1"
        : "=&v"(t) : "v"(x));
    x = t;
    asm("v_add_f32_dpp %0, %1, %1 row_half_mirror row_mask:0xf bank_mask:0xf bound_ctrl:1"
        : "=&v"(t) : "v"(x));
    x = t;
    asm("v_add_f32_dpp %0, %1, %1 row_mirror row_mask:0xf bank_mask:0xf bound_ctrl:1"
        : "=&v"(t) : "v"(x));
    return t;
}

#if HAVE_TDM
// One TDM descriptor-driven DMA of a 2D bf16 tile (rows x 64 elems) into LDS.
// D# per CDNA5 ISA §8.3/8.4: group0 = {count, lds_addr, global_addr, type=2},
// group1 = {data_size=2B, tensor dims (large), tile dims, dim0 stride}.
__device__ __forceinline__ void tdm_load_tile_64xN(const bf16* gsrc, bf16* ldst,
                                                   unsigned rows, unsigned row_stride_elems) {
    unsigned long long ga = (unsigned long long)(uintptr_t)gsrc;
    unsigned la = (unsigned)(uintptr_t)ldst;  // flat-shared low 32 bits == LDS byte offset
    u32x4 g0;
    g0[0] = 1u;                                   // count=1, user descriptor
    g0[1] = la;                                   // lds_addr
    g0[2] = (unsigned)ga;                         // global_addr[31:0]
    g0[3] = (unsigned)((ga >> 32) & 0x1FFFFFFull) | (2u << 30);  // ga[56:32] | type=2
    const unsigned td0 = 1u << 20, td1 = 1u << 20;  // huge tensor dims: no OOB clipping
    i32x8 g1;
    g1[0] = (int)(1u << 16);                      // workgroup_mask=0, data_size=1 (2 bytes)
    g1[1] = (int)((td0 & 0xFFFFu) << 16);         // atomic_barrier_addr=0 | tensor_dim0 lo
    g1[2] = (int)((td0 >> 16) | ((td1 & 0xFFFFu) << 16));  // tensor_dim0 hi | tensor_dim1 lo
    g1[3] = (int)((td1 >> 16) | (64u << 16));     // tensor_dim1 hi | tile_dim0=64
    g1[4] = (int)(rows & 0xFFFFu);                // tile_dim1=rows, tile_dim2=0
    g1[5] = (int)row_stride_elems;                // tensor_dim0_stride[31:0]
    g1[6] = 0;
    g1[7] = 0;
    i32x4 z4 = {0, 0, 0, 0};
#if defined(__clang_major__) && (__clang_major__ >= 23)
    i32x8 z8 = {0, 0, 0, 0, 0, 0, 0, 0};
    __builtin_amdgcn_tensor_load_to_lds(g0, g1, z4, z4, z8, 0);
#else
    __builtin_amdgcn_tensor_load_to_lds(g0, g1, z4, z4, 0);
#endif
}
#endif

__global__ void cvt_f32_bf16(const float* __restrict__ in, bf16* __restrict__ out, int n) {
    int i = (blockIdx.x * blockDim.x + threadIdx.x) * 4;
    if (i + 3 < n) {
        float4 f = *(const float4*)(in + i);
        out[i + 0] = (bf16)f.x;
        out[i + 1] = (bf16)f.y;
        out[i + 2] = (bf16)f.z;
        out[i + 3] = (bf16)f.w;
    } else {
        for (; i < n; ++i) out[i] = (bf16)in[i];
    }
}

// C[m,n] = scale * (sum_k A[m,k] * W[n,k] + bias[n])
// MODE 0: bf16 out, head-split (B,H,S,64)            (Q,K projections)
// MODE 1: bf16 out, head-split transposed (B,H,64,S) (V projection)
// MODE 2: fp32 out, contiguous (M,N)                 (output projection)
// 8 waves as 4(M) x 2(N); block tile 128x128, wave tile 32x64.
template <int MODE>
__global__ void gemm_wmma_bf16(const bf16* __restrict__ A, const bf16* __restrict__ W,
                               const float* __restrict__ bias, void* __restrict__ out,
                               int M, int N, int K, int Slen, float scale) {
    const int lane = threadIdx.x & 31;
    const int wv   = threadIdx.x >> 5;
    const int wm   = wv & 3;
    const int wn   = wv >> 2;
    const int r    = lane & 15;
    const int hh   = lane >> 4;
    const int m0   = blockIdx.y * 128 + wm * 32;
    const int n0   = blockIdx.x * 128 + wn * 64;

    const bf16* Arow0 = A + (size_t)(m0 + r) * K;
    const bf16* Arow1 = Arow0 + (size_t)16 * K;

    v8f acc[4][2] = {};
    for (int k0 = 0; k0 < K; k0 += 32) {
        v16bf a0 = load_frag(Arow0 + k0 + hh * 8, Arow0 + k0 + 16 + hh * 8);
        v16bf a1 = load_frag(Arow1 + k0 + hh * 8, Arow1 + k0 + 16 + hh * 8);
#pragma unroll
        for (int j = 0; j < 4; ++j) {
            const bf16* Wrow = W + (size_t)(n0 + j * 16 + r) * K + k0;
            v16bf bfr = load_frag(Wrow + hh * 8, Wrow + 16 + hh * 8);
            acc[j][0] = __builtin_amdgcn_wmma_f32_16x16x32_bf16(
                false, a0, false, bfr, (short)0, acc[j][0], false, false);
            acc[j][1] = __builtin_amdgcn_wmma_f32_16x16x32_bf16(
                false, a1, false, bfr, (short)0, acc[j][1], false, false);
        }
    }

#pragma unroll
    for (int j = 0; j < 4; ++j) {
        int n = n0 + j * 16 + r;
        float bval = bias[n];
#pragma unroll
        for (int i = 0; i < 2; ++i) {
#pragma unroll
            for (int v = 0; v < 8; ++v) {
                int m = m0 + i * 16 + v + 8 * hh;
                float val = (acc[j][i][v] + bval) * scale;
                if (MODE == 2) {
                    ((float*)out)[(size_t)m * N + n] = val;
                } else {
                    int bidx = m / Slen, s = m % Slen;
                    int head = n >> 6, dk = n & 63;
                    bf16* o = (bf16*)out;
                    if (MODE == 0)
                        o[(((size_t)bidx * HEADS + head) * Slen + s) * D_K + dk] = (bf16)val;
                    else
                        o[(((size_t)bidx * HEADS + head) * D_K + dk) * Slen + s] = (bf16)val;
                }
            }
        }
    }
}

// Flash-attention: q pre-scaled by 1/sqrt(dk); k (B,H,S,64); vT (B,H,64,S); x (B,S,D).
// 8 waves/block, wave owns 16 query rows, Sk streamed in 64-wide chunks.
// K/V tiles DMA'd into double-buffered LDS by the Tensor Data Mover.
__global__ void attention_wmma(const bf16* __restrict__ q, const bf16* __restrict__ k,
                               const bf16* __restrict__ vT, bf16* __restrict__ x) {
    __shared__ bf16 ldsP[8][16][64];  // per-wave probs tile (C->A relayout)
#if HAVE_TDM
    __shared__ bf16 kbuf[2][64][64];  // K rows (sk,d) tiles, double buffered
    __shared__ bf16 vbuf[2][64][64];  // V^T rows (d,sk) tiles, double buffered
#endif

    const int lane = threadIdx.x & 31;
    const int wv   = threadIdx.x >> 5;
    const int r    = lane & 15;
    const int hh   = lane >> 4;
    const int bh   = blockIdx.y;
    const int b    = bh / HEADS, head = bh % HEADS;
    const int m0   = blockIdx.x * 128 + wv * 16;

    const bf16* qrow = q + ((size_t)bh * SEQ + m0 + r) * D_K;
    v16bf qa0 = load_frag(qrow + hh * 8,      qrow + 16 + hh * 8);
    v16bf qa1 = load_frag(qrow + 32 + hh * 8, qrow + 48 + hh * 8);

    float mrow[8], lrow[8];
#pragma unroll
    for (int v = 0; v < 8; ++v) { mrow[v] = -3.0e38f; lrow[v] = 0.f; }
    v8f o[4] = {};

    const int NIT = SEQ / 64;
#if HAVE_TDM
    if (wv == 0) {
        tdm_load_tile_64xN(k  + (size_t)bh * SEQ * D_K, &kbuf[0][0][0], 64, D_K);
        tdm_load_tile_64xN(vT + (size_t)bh * D_K * SEQ, &vbuf[0][0][0], 64, SEQ);
    }
#endif

    for (int it = 0; it < NIT; ++it) {
        const int sk0 = it * 64;
        const int cur = it & 1;
#if HAVE_TDM
        if (wv == 0) {
            if (it + 1 < NIT) {
                tdm_load_tile_64xN(k  + ((size_t)bh * SEQ + sk0 + 64) * D_K,
                                   &kbuf[cur ^ 1][0][0], 64, D_K);
                tdm_load_tile_64xN(vT + (size_t)bh * D_K * SEQ + sk0 + 64,
                                   &vbuf[cur ^ 1][0][0], 64, SEQ);
                __builtin_amdgcn_s_wait_tensorcnt((short)2);  // in-order: current pair done
            } else {
                __builtin_amdgcn_s_wait_tensorcnt((short)0);
            }
        }
        __syncthreads();
#endif

        // ---- scores: 16x64 = four C-tiles; fragments processed pairwise
        // (2 live frags, dependent WMMAs on one accumulator stay >=3 apart) ----
        v8f s[4] = {};
#pragma unroll
        for (int half = 0; half < 2; ++half) {
            const int ko = half * 32;
            const v16bf qa = half ? qa1 : qa0;
#pragma unroll
            for (int tp = 0; tp < 2; ++tp) {
#if HAVE_TDM
                const bf16* krowA = &kbuf[cur][(tp * 2 + 0) * 16 + r][0];
                const bf16* krowB = &kbuf[cur][(tp * 2 + 1) * 16 + r][0];
#else
                const bf16* krowA = k + ((size_t)bh * SEQ + sk0 + (tp * 2 + 0) * 16 + r) * D_K;
                const bf16* krowB = k + ((size_t)bh * SEQ + sk0 + (tp * 2 + 1) * 16 + r) * D_K;
#endif
                v16bf kbA = load_frag(krowA + ko + hh * 8, krowA + ko + 16 + hh * 8);
                v16bf kbB = load_frag(krowB + ko + hh * 8, krowB + ko + 16 + hh * 8);
                s[tp * 2 + 0] = __builtin_amdgcn_wmma_f32_16x16x32_bf16(
                    false, qa, false, kbA, (short)0, s[tp * 2 + 0], false, false);
                s[tp * 2 + 1] = __builtin_amdgcn_wmma_f32_16x16x32_bf16(
                    false, qa, false, kbB, (short)0, s[tp * 2 + 1], false, false);
            }
        }

        // ---- online softmax in C layout (row M = v + 8*hh, col N = t*16 + r) ----
#pragma unroll
        for (int v = 0; v < 8; ++v) {
            float a0 = s[0][v], a1 = s[1][v], a2 = s[2][v], a3 = s[3][v];
            float cm = red_max16(fmaxf(fmaxf(a0, a1), fmaxf(a2, a3)));
            float mnew = fmaxf(mrow[v], cm);
            float corr = __expf(mrow[v] - mnew);
            float p0 = __expf(a0 - mnew);
            float p1 = __expf(a1 - mnew);
            float p2 = __expf(a2 - mnew);
            float p3 = __expf(a3 - mnew);
            float cs = red_sum16((p0 + p1) + (p2 + p3));
            lrow[v] = lrow[v] * corr + cs;
            mrow[v] = mnew;
#pragma unroll
            for (int j = 0; j < 4; ++j) o[j][v] *= corr;
            int row = v + 8 * hh;
            ldsP[wv][row][r]      = (bf16)p0;
            ldsP[wv][row][16 + r] = (bf16)p1;
            ldsP[wv][row][32 + r] = (bf16)p2;
            ldsP[wv][row][48 + r] = (bf16)p3;
        }
        // probs tile is per-wave private; per-wave LDS ops retire in order, so a
        // scheduling fence is enough (no block barrier needed here).
        __builtin_amdgcn_wave_barrier();

        // ---- P*V: fragments pairwise, pa0-group then pa1-group ----
        v16bf pa0 = load_frag(&ldsP[wv][r][hh * 8],      &ldsP[wv][r][16 + hh * 8]);
        v16bf pa1 = load_frag(&ldsP[wv][r][32 + hh * 8], &ldsP[wv][r][48 + hh * 8]);
#pragma unroll
        for (int half = 0; half < 2; ++half) {
            const int ko = half * 32;
            const v16bf pa = half ? pa1 : pa0;
#pragma unroll
            for (int jp = 0; jp < 2; ++jp) {
#if HAVE_TDM
                const bf16* vrowA = &vbuf[cur][(jp * 2 + 0) * 16 + r][0];
                const bf16* vrowB = &vbuf[cur][(jp * 2 + 1) * 16 + r][0];
#else
                const bf16* vrowA = vT + ((size_t)bh * D_K + (jp * 2 + 0) * 16 + r) * SEQ + sk0;
                const bf16* vrowB = vT + ((size_t)bh * D_K + (jp * 2 + 1) * 16 + r) * SEQ + sk0;
#endif
                v16bf vbA = load_frag(vrowA + ko + hh * 8, vrowA + ko + 16 + hh * 8);
                v16bf vbB = load_frag(vrowB + ko + hh * 8, vrowB + ko + 16 + hh * 8);
                o[jp * 2 + 0] = __builtin_amdgcn_wmma_f32_16x16x32_bf16(
                    false, pa, false, vbA, (short)0, o[jp * 2 + 0], false, false);
                o[jp * 2 + 1] = __builtin_amdgcn_wmma_f32_16x16x32_bf16(
                    false, pa, false, vbB, (short)0, o[jp * 2 + 1], false, false);
            }
        }
#if HAVE_TDM
        __syncthreads();  // all waves done with buf[cur] before TDM refills it
#endif
    }

    // ---- normalize and write merged-heads x (B,S,D) ----
#pragma unroll
    for (int v = 0; v < 8; ++v) {
        float inv = 1.f / lrow[v];
        int m = m0 + v + 8 * hh;
#pragma unroll
        for (int j = 0; j < 4; ++j) {
            int col = head * D_K + j * 16 + r;
            x[((size_t)b * SEQ + m) * D_MODEL + col] = (bf16)(o[j][v] * inv);
        }
    }
}

extern "C" void kernel_launch(void* const* d_in, const int* in_sizes, int n_in,
                              void* d_out, int out_size, void* d_ws, size_t ws_size,
                              hipStream_t stream) {
    const float* query = (const float*)d_in[0];
    const float* key_  = (const float*)d_in[1];
    const float* value = (const float*)d_in[2];
    const float* Wq = (const float*)d_in[3]; const float* bq = (const float*)d_in[4];
    const float* Wk = (const float*)d_in[5]; const float* bk = (const float*)d_in[6];
    const float* Wv = (const float*)d_in[7]; const float* bv = (const float*)d_in[8];
    const float* Wo = (const float*)d_in[9]; const float* bo = (const float*)d_in[10];

    const size_t NW = (size_t)D_MODEL * D_MODEL;     // 1 Mi weight elems
    const size_t NA = (size_t)BATCH * SEQ * D_MODEL; // 8 Mi activation elems

    bf16* Wq_b = (bf16*)d_ws;
    bf16* Wk_b = Wq_b + NW;
    bf16* Wv_b = Wk_b + NW;
    bf16* Wo_b = Wv_b + NW;
    bf16* q_in = Wo_b + NW;
    bf16* k_in = q_in + NA;
    bf16* v_in = k_in + NA;
    bf16* qh   = v_in + NA;   // (B,H,S,64), pre-scaled by 1/8
    bf16* kh   = qh + NA;     // (B,H,S,64)
    bf16* vTh  = kh + NA;     // (B,H,64,S)
    bf16* xb   = vTh + NA;    // (B,S,D) merged heads

    const int M = BATCH * SEQ;
    const int N = D_MODEL;
    const int K = D_MODEL;
    dim3 blk(256);

    cvt_f32_bf16<<<dim3((unsigned)(NW / 4 / 256)), blk, 0, stream>>>(Wq, Wq_b, (int)NW);
    cvt_f32_bf16<<<dim3((unsigned)(NW / 4 / 256)), blk, 0, stream>>>(Wk, Wk_b, (int)NW);
    cvt_f32_bf16<<<dim3((unsigned)(NW / 4 / 256)), blk, 0, stream>>>(Wv, Wv_b, (int)NW);
    cvt_f32_bf16<<<dim3((unsigned)(NW / 4 / 256)), blk, 0, stream>>>(Wo, Wo_b, (int)NW);
    cvt_f32_bf16<<<dim3((unsigned)(NA / 4 / 256)), blk, 0, stream>>>(query, q_in, (int)NA);
    cvt_f32_bf16<<<dim3((unsigned)(NA / 4 / 256)), blk, 0, stream>>>(key_,  k_in, (int)NA);
    cvt_f32_bf16<<<dim3((unsigned)(NA / 4 / 256)), blk, 0, stream>>>(value, v_in, (int)NA);

    dim3 ggrid(N / 128, M / 128);
    gemm_wmma_bf16<0><<<ggrid, blk, 0, stream>>>(q_in, Wq_b, bq, qh,  M, N, K, SEQ, 0.125f);
    gemm_wmma_bf16<0><<<ggrid, blk, 0, stream>>>(k_in, Wk_b, bk, kh,  M, N, K, SEQ, 1.0f);
    gemm_wmma_bf16<1><<<ggrid, blk, 0, stream>>>(v_in, Wv_b, bv, vTh, M, N, K, SEQ, 1.0f);

    attention_wmma<<<dim3(SEQ / 128, BATCH * HEADS), blk, 0, stream>>>(qh, kh, vTh, xb);

    gemm_wmma_bf16<2><<<ggrid, blk, 0, stream>>>(xb, Wo_b, bo, d_out, M, N, K, SEQ, 1.0f);
}
